// ProbAttention_17549236371540
// MI455X (gfx1250) — matrix operand order repaired and checked by reading, not compiled, not run on hardware
//
#include <hip/hip_runtime.h>
#include <hip/hip_bf16.h>

// ---------------------------------------------------------------------------
// ProbSparse attention (Informer) for MI455X / gfx1250.
// B=4 H=8 D=64 L=4096, FACTOR=5 -> U_part=25 sampled keys, u=45 top queries.
// Output = cumsum(values,axis=L)/64 with 45 columns per (b,h) overwritten by
// a 48x4096 flash-softmax attention computed with v_wmma_f32_16x16x32_f16.
// Memory floor: 96MB read + 32MB write @ 23.3 TB/s ~= 5.8us; compute is
// ~2 GFLOP -> utterly bandwidth-bound, so the cumsum kernel uses b128
// loads/stores and a minimal-barrier block scan.
// ---------------------------------------------------------------------------

typedef __attribute__((ext_vector_type(16))) _Float16 v16h;
typedef __attribute__((ext_vector_type(8)))  float    v8f;

union AFrag { v16h v; _Float16 h[16]; };
union CFrag { v8f  v; float    f[8];  };

#define L_LEN 4096
#define D_LEN 64
#define NSAMP 25
#define NTOP  45
#define NBH   32

// ---------------- JAX threefry2x32 (key = (0, 42)) -------------------------
__device__ __forceinline__ unsigned tf_rotl(unsigned x, int r) {
    return (x << r) | (x >> (32 - r));
}

__device__ __forceinline__ void threefry2x32(unsigned k0, unsigned k1,
                                             unsigned x0, unsigned x1,
                                             unsigned& y0, unsigned& y1) {
    unsigned ks0 = k0, ks1 = k1, ks2 = k0 ^ k1 ^ 0x1BD11BDAu;
    const int rotA[4] = {13, 15, 26, 6};
    const int rotB[4] = {17, 29, 16, 24};
    x0 += ks0; x1 += ks1;
#pragma unroll
    for (int g = 0; g < 5; ++g) {
        const int* rr = (g & 1) ? rotB : rotA;
#pragma unroll
        for (int r = 0; r < 4; ++r) {
            x0 += x1; x1 = tf_rotl(x1, rr[r]); x1 ^= x0;
        }
        unsigned ks[3] = {ks0, ks1, ks2};
        x0 += ks[(g + 1) % 3];
        x1 += ks[(g + 2) % 3] + (unsigned)(g + 1);
    }
    y0 = x0; y1 = x1;
}

// index_sample[l][s]: flat i = l*25+s; counts 0..102399 split into halves.
__device__ __forceinline__ int sample_index(int l, int s) {
    unsigned i = (unsigned)(l * NSAMP + s);
    unsigned j = (i < 51200u) ? i : i - 51200u;
    unsigned y0, y1;
    threefry2x32(0u, 42u, j, j + 51200u, y0, y1);
    unsigned bits = (i < 51200u) ? y0 : y1;
    return (int)(bits & 0xFFFu);   // span 4096 is a power of two
}

// ---------------- Kernel 1: sparsity measure M ------------------------------
__global__ void k_sample_M(const float* __restrict__ q,
                           const float* __restrict__ k,
                           float* __restrict__ Mout) {
    int l  = blockIdx.x * blockDim.x + threadIdx.x;   // query position
    int bh = blockIdx.y;
    const float* qb = q + (size_t)bh * D_LEN * L_LEN;
    const float* kb = k + (size_t)bh * D_LEN * L_LEN;

    float qv[D_LEN];
#pragma unroll
    for (int d = 0; d < D_LEN; ++d) qv[d] = qb[(size_t)d * L_LEN + l]; // coalesced

    float mx = -1e30f, sm = 0.f;
    for (int s = 0; s < NSAMP; ++s) {
        int idx = sample_index(l, s);
        float dot = 0.f;
#pragma unroll
        for (int d = 0; d < D_LEN; ++d) dot += qv[d] * kb[(size_t)d * L_LEN + idx];
        mx = fmaxf(mx, dot);
        sm += dot;
    }
    const float ds = (1.0f / 512.0f) * (1.0f / 64.0f);  // q,k pre-scales
    Mout[(size_t)bh * L_LEN + l] = (mx - sm * (1.0f / 4096.0f)) * ds;
}

// ---------------- Kernel 2: top-45 per (b,h) --------------------------------
__global__ void k_topk(const float* __restrict__ M, int* __restrict__ top) {
    __shared__ float sm[L_LEN];
    __shared__ float rv[256];
    __shared__ int   ri[256];
    int t = threadIdx.x, bh = blockIdx.x;
    for (int i = t; i < L_LEN; i += 256) sm[i] = M[(size_t)bh * L_LEN + i];
    __syncthreads();
    for (int it = 0; it < NTOP; ++it) {
        float bv = -1e38f; int bi = 0;
        for (int i = t; i < L_LEN; i += 256) {
            float v = sm[i];
            if (v > bv || (v == bv && i < bi)) { bv = v; bi = i; }
        }
        rv[t] = bv; ri[t] = bi;
        __syncthreads();
        for (int off = 128; off > 0; off >>= 1) {
            if (t < off) {
                float v2 = rv[t + off]; int i2 = ri[t + off];
                if (v2 > rv[t] || (v2 == rv[t] && i2 < ri[t])) { rv[t] = v2; ri[t] = i2; }
            }
            __syncthreads();
        }
        if (t == 0) { top[bh * NTOP + it] = ri[0]; sm[ri[0]] = -1e38f; }
        __syncthreads();
    }
}

// ---------------- Kernel 3: streaming prefix sum (the bandwidth kernel) ----
// One (b,h,d) row (4096 floats) per block. Each thread owns one float4 per
// chunk (global_load_b128 / global_store_b128), serial-scans it in registers,
// and only the 256 per-thread totals go through the LDS scan: 4 chunks x 8
// rounds instead of 16 x 8, all VMEM 128-bit.
__global__ void k_cumsum(const float4* __restrict__ v4, float4* __restrict__ out4) {
    __shared__ float tsum[256];
    int t = threadIdx.x;
    size_t base = (size_t)blockIdx.x * (L_LEN / 4);   // row base, float4 units
    float carry = 0.f;
#pragma unroll
    for (int c = 0; c < 4; ++c) {
        size_t i = base + c * 256 + t;
        float4 x = v4[i];
        if (c + 1 < 4) __builtin_prefetch(&v4[i + 256], 0, 0); // global_prefetch_b8
        // serial inclusive scan of the 4 owned elements
        float s0 = x.x;
        float s1 = s0 + x.y;
        float s2 = s1 + x.z;
        float s3 = s2 + x.w;
        tsum[t] = s3;
        __syncthreads();
        for (int off = 1; off < 256; off <<= 1) {     // Hillis-Steele on totals
            float y = (t >= off) ? tsum[t - off] : 0.f;
            __syncthreads();
            tsum[t] += y;
            __syncthreads();
        }
        float excl = carry + ((t > 0) ? tsum[t - 1] : 0.f);
        float tot  = tsum[255];
        __syncthreads();   // protect tsum before next chunk's overwrite
        const float inv = 1.0f / 64.0f;               // fold v/64
        float4 o;
        o.x = (excl + s0) * inv;
        o.y = (excl + s1) * inv;
        o.z = (excl + s2) * inv;
        o.w = (excl + s3) * inv;
        out4[i] = o;
        carry += tot;
    }
}

// ---------------- Kernel 4: WMMA flash attention for 45 selected rows -------
// One wave per (m-tile, bh). M-tile = 16 rows (3 tiles cover 45, padded to 48).
// Key blocks of 32: S(16x32) via 4x v_wmma_f32_16x16x32_f16, online softmax,
// P routed through LDS to A-layout, O(16x64) += P@V via 4 more WMMAs.
__global__ void __launch_bounds__(32)
k_attn(const float* __restrict__ q, const float* __restrict__ k,
       const float* __restrict__ v, const int* __restrict__ top,
       float* __restrict__ out) {
    __shared__ int   qidx_lds[16];
    __shared__ float p_lds[16 * 32];

    int lane  = threadIdx.x;
    int mtile = blockIdx.x;
    int bh    = blockIdx.y;
    int hi    = lane >> 4;        // half-wave
    int n     = lane & 15;
    size_t bbase = (size_t)bh * D_LEN * L_LEN;

    if (lane < 16) {
        int u = mtile * 16 + lane;
        if (u > NTOP - 1) u = NTOP - 1;            // pad rows duplicate row 44
        qidx_lds[lane] = top[bh * NTOP + u];
    }
    __syncthreads();

    // A-fragments of Q_reduce: lane holds row (lane&15); 16-bit A layout:
    // e<8 -> K = e + 8*hi ; e>=8 -> K = e + 8 + 8*hi  (per 16x32 ISA table)
    AFrag aq[2];
    int row  = n;
    int qcol = qidx_lds[row];
#pragma unroll
    for (int ks = 0; ks < 2; ++ks)
#pragma unroll
        for (int e = 0; e < 16; ++e) {
            int kk = (e < 8) ? (e + 8 * hi) : (e + 8 + 8 * hi);
            int d  = 32 * ks + kk;
            aq[ks].h[e] = (_Float16)q[bbase + (size_t)d * L_LEN + qcol];
        }

    float rmax[8], rsum[8];
    CFrag O[4];
#pragma unroll
    for (int i = 0; i < 8; ++i) { rmax[i] = -1e30f; rsum[i] = 0.f; }
#pragma unroll
    for (int tt = 0; tt < 4; ++tt)
#pragma unroll
        for (int i = 0; i < 8; ++i) O[tt].f[i] = 0.f;

    const float SCALE = 1.0f / (512.0f * 64.0f * 512.0f);  // 2^-24 combined

    for (int kb = 0; kb < L_LEN / 32; ++kb) {
        int l0 = kb * 32;

        // ---- S = Qr @ K^T for 32 keys (two 16x16 C tiles) ----
        CFrag S[2];
#pragma unroll
        for (int nh = 0; nh < 2; ++nh) {
#pragma unroll
            for (int i = 0; i < 8; ++i) S[nh].f[i] = 0.f;
#pragma unroll
            for (int ks = 0; ks < 2; ++ks) {
                AFrag bk;   // B layout: e<8 -> K = e + 8*hi ; e>=8 -> K = e + 8 + 8*hi, column = n
#pragma unroll
                for (int e = 0; e < 16; ++e) {
                    int d = 32 * ks + ((e < 8) ? (e + 8 * hi) : (e + 8 + 8 * hi));
                    bk.h[e] = (_Float16)k[bbase + (size_t)d * L_LEN + (l0 + 16 * nh + n)];
                }
                S[nh].v = __builtin_amdgcn_wmma_f32_16x16x32_f16(
                    false, aq[ks].v, false, bk.v, (short)0, S[nh].v, false, false);
            }
        }

        // ---- scale + causal mask + online softmax ----
        float bmax[8];
#pragma unroll
        for (int i = 0; i < 8; ++i) {
            int m  = i + 8 * hi;            // C layout: row = i + 8*hi, col = n
            int qi = qidx_lds[m];
#pragma unroll
            for (int nh = 0; nh < 2; ++nh) {
                int   l  = l0 + 16 * nh + n;
                float sv = S[nh].f[i] * SCALE;
                if (l > qi) sv = -1e30f;    // sentinel: l=0 always unmasked
                S[nh].f[i] = sv;
            }
            float bm = fmaxf(S[0].f[i], S[1].f[i]);
#pragma unroll
            for (int off = 1; off < 16; off <<= 1)
                bm = fmaxf(bm, __shfl_xor(bm, off, 32));
            bmax[i] = bm;
        }
#pragma unroll
        for (int i = 0; i < 8; ++i) {
            float mn   = fmaxf(rmax[i], bmax[i]);
            float corr = __expf(rmax[i] - mn);
            rmax[i] = mn;
            float bs = 0.f;
#pragma unroll
            for (int nh = 0; nh < 2; ++nh) {
                float p = __expf(S[nh].f[i] - mn);
                S[nh].f[i] = p;
                bs += p;
            }
#pragma unroll
            for (int off = 1; off < 16; off <<= 1)
                bs += __shfl_xor(bs, off, 32);
            rsum[i] = rsum[i] * corr + bs;
#pragma unroll
            for (int tt = 0; tt < 4; ++tt) O[tt].f[i] *= corr;
        }

        // ---- P: C layout -> A layout via LDS ----
        __syncthreads();
#pragma unroll
        for (int i = 0; i < 8; ++i) {
            int m = i + 8 * hi;
            p_lds[m * 32 + n]      = S[0].f[i];
            p_lds[m * 32 + 16 + n] = S[1].f[i];
        }
        __syncthreads();
        AFrag ap;
#pragma unroll
        for (int e = 0; e < 16; ++e) {
            int kk = (e < 8) ? (e + 8 * hi) : (e + 8 + 8 * hi);
            ap.h[e] = (_Float16)p_lds[row * 32 + kk];
        }

        // ---- O(16x64) += P(16x32) @ V(32x64) ----
#pragma unroll
        for (int tt = 0; tt < 4; ++tt) {
            AFrag bv;
#pragma unroll
            for (int e = 0; e < 16; ++e) {
                int kk = (e < 8) ? (e + 8 * hi) : (e + 8 + 8 * hi);
                bv.h[e] = (_Float16)v[bbase + (size_t)(tt * 16 + n) * L_LEN + l0 + kk];
            }
            O[tt].v = __builtin_amdgcn_wmma_f32_16x16x32_f16(
                false, ap.v, false, bv.v, (short)0, O[tt].v, false, false);
        }
    }

    // ---- normalize and scatter into the cumsum output ----
#pragma unroll
    for (int i = 0; i < 8; ++i) {
        int m = i + 8 * hi;
        int u = mtile * 16 + m;
        if (u < NTOP) {
            int   col = qidx_lds[m];
            float inv = 1.0f / (rsum[i] * 64.0f);  // fold v/64
#pragma unroll
            for (int tt = 0; tt < 4; ++tt) {
                int d = tt * 16 + n;
                out[bbase + (size_t)d * L_LEN + col] = O[tt].f[i] * inv;
            }
        }
    }
}

// ---------------------------------------------------------------------------
extern "C" void kernel_launch(void* const* d_in, const int* in_sizes, int n_in,
                              void* d_out, int out_size, void* d_ws, size_t ws_size,
                              hipStream_t stream) {
    const float* q = (const float*)d_in[0];
    const float* k = (const float*)d_in[1];
    const float* v = (const float*)d_in[2];
    float* out = (float*)d_out;

    // workspace: M values (32*4096 f32 = 512KB) + top indices (32*45 i32)
    float* Mbuf   = (float*)d_ws;
    int*   topbuf = (int*)((char*)d_ws + (size_t)NBH * L_LEN * sizeof(float));

    k_sample_M<<<dim3(L_LEN / 256, NBH), 256, 0, stream>>>(q, k, Mbuf);
    k_topk<<<NBH, 256, 0, stream>>>(Mbuf, topbuf);
    k_cumsum<<<NBH * D_LEN, 256, 0, stream>>>((const float4*)v, (float4*)out);
    k_attn<<<dim3(3, NBH), 32, 0, stream>>>(q, k, v, topbuf, out);
}
